// model_st_7060926234913
// MI455X (gfx1250) — matrix operand order repaired and checked by reading, not compiled, not run on hardware
//
#include <hip/hip_runtime.h>
#include <hip/hip_bf16.h>
#include <math.h>

// ---------------------------------------------------------------------------
// Types for CDNA5 WMMA (wave32, v_wmma_f32_16x16x32_bf16)
// ---------------------------------------------------------------------------
typedef __attribute__((ext_vector_type(16))) __bf16 v16bf;
typedef __attribute__((ext_vector_type(8)))  __bf16 v8bf;
typedef __attribute__((ext_vector_type(8)))  float  v8f;

#define NUM_GRAPHS 2048
#define NODES_PER_GRAPH 50
#define NUM_NODES (NUM_GRAPHS * NODES_PER_GRAPH)
#define NUM_EDGES (NUM_GRAPHS * 150)
#define NUM_SE 994
#define SE_FEAT 627
#define NUM_SE_EDGES 20000

__device__ __forceinline__ float activate(float v, int act) {
    switch (act) {
        case 1: return v > 0.0f ? v : 0.0f;          // relu
        case 2: return tanhf(v);                      // tanh
        case 3: return v > 0.0f ? v : 0.01f * v;      // leaky relu
        default: return v;
    }
}

// Build a v16bf fragment from two 16B-aligned contiguous LDS slices.
__device__ __forceinline__ v16bf frag16(const __bf16* p0, const __bf16* p1) {
    v8bf lo = *(const v8bf*)p0;
    v8bf hi = *(const v8bf*)p1;
    v16bf r;
    #pragma unroll
    for (int i = 0; i < 8; ++i) { r[i] = lo[i]; r[i + 8] = hi[i]; }
    return r;
}

// ---------------------------------------------------------------------------
// Generic tiled GEMM: C[M,N] = act(A[M,K] @ B + bias), B is [K,N] row-major
// (transB==0) or [N,K] row-major (transB==1, i.e. C = A @ B^T).
// bf16 WMMA, f32 accumulate. Block tile 128x64, 8 waves arranged 4(M) x 2(N),
// each wave owns a 32x32 slice: 2 A-frags x 2 B-frags -> 4 WMMAs per K-step.
// Global loads are unconditional in the steady state (row/col indices are
// clamped; OOB data only reaches never-stored outputs). Only the final
// partial-K tile takes the guarded path.
// ---------------------------------------------------------------------------
#define TM 128
#define TN 64
#define TK 32
#define LDSS 40   // padded LDS row stride in bf16 elems (80B, 16B aligned)

__global__ __launch_bounds__(256) void gemm_wmma_bf16(
    const float* __restrict__ A, const float* __restrict__ B,
    const float* __restrict__ bias, float* __restrict__ C,
    int M, int N, int K, int ldc, int transB, int act)
{
    __shared__ __align__(16) __bf16 As[TM * LDSS];  // [row][k]
    __shared__ __align__(16) __bf16 Bs[TN * LDSS];  // [n][k] (transposed)

    const int m0 = blockIdx.y * TM;
    const int n0 = blockIdx.x * TN;
    const int t    = threadIdx.x;
    const int lane = t & 31;
    const int w    = t >> 5;
    const int wm   = (w & 3) * 32;   // wave row offset inside tile
    const int wn   = (w >> 2) * 32;  // wave col offset inside tile

    v8f acc[2][2] = {};

    // A-load mapping: k-col fixed per thread (coalesced), rows r0 + 8*i
    const int ac  = t & 31;
    const int ar0 = t >> 5;          // 0..7
    // B-load mapping (transB==0): n fixed per thread (coalesced), k = c0 + 4*i
    const int bn  = t & 63;
    const int bc0 = t >> 6;          // 0..3
    // B-load mapping (transB==1): k fixed per thread (coalesced), n = n0b + 8*i
    const int bc  = t & 31;
    const int bn0 = t >> 5;          // 0..7

    const int kt = (K + TK - 1) / TK;
    for (int kk = 0; kk < kt; ++kk) {
        const int k0 = kk * TK;
        float va[16];
        float vb[8];
        if (k0 + TK <= K) {
            // ---------- steady state: no k guards, clamped rows/cols ----------
            #pragma unroll
            for (int i = 0; i < 16; ++i) {
                int gr = m0 + ar0 + i * 8; gr = gr < M ? gr : M - 1;
                va[i] = A[(long)gr * K + k0 + ac];
            }
            if (!transB) {
                int gn = n0 + bn; gn = gn < N ? gn : N - 1;
                #pragma unroll
                for (int i = 0; i < 8; ++i)
                    vb[i] = B[(long)(k0 + bc0 + i * 4) * N + gn];
            } else {
                #pragma unroll
                for (int i = 0; i < 8; ++i) {
                    int gn = n0 + bn0 + i * 8; gn = gn < N ? gn : N - 1;
                    vb[i] = B[(long)gn * K + k0 + bc];
                }
            }
        } else {
            // ---------- final partial K tile: guard k ----------
            #pragma unroll
            for (int i = 0; i < 16; ++i) {
                int gr = m0 + ar0 + i * 8; gr = gr < M ? gr : M - 1;
                va[i] = (k0 + ac < K) ? A[(long)gr * K + k0 + ac] : 0.0f;
            }
            if (!transB) {
                int gn = n0 + bn; gn = gn < N ? gn : N - 1;
                #pragma unroll
                for (int i = 0; i < 8; ++i) {
                    int gc = k0 + bc0 + i * 4;
                    vb[i] = (gc < K) ? B[(long)gc * N + gn] : 0.0f;
                }
            } else {
                #pragma unroll
                for (int i = 0; i < 8; ++i) {
                    int gn = n0 + bn0 + i * 8; gn = gn < N ? gn : N - 1;
                    vb[i] = (k0 + bc < K) ? B[(long)gn * K + k0 + bc] : 0.0f;
                }
            }
        }
        // ---- convert + stage to LDS ----
        #pragma unroll
        for (int i = 0; i < 16; ++i)
            As[(ar0 + i * 8) * LDSS + ac] = (__bf16)va[i];
        if (!transB) {
            #pragma unroll
            for (int i = 0; i < 8; ++i)
                Bs[bn * LDSS + bc0 + i * 4] = (__bf16)vb[i];
        } else {
            #pragma unroll
            for (int i = 0; i < 8; ++i)
                Bs[(bn0 + i * 8) * LDSS + bc] = (__bf16)vb[i];
        }
        __syncthreads();

        // ---- fragments (ISA 16-bit A 16x32 / B 32x16 lane layouts) ----
        const int fr    = lane & 15;
        const int khalf = (lane >> 4) * 8;   // A: elems 0-7 -> khalf.., 8-15 -> 16+khalf..
        const int kb    = (lane >> 4) * 16;  // B: same k order as A (khalf.., 16+khalf..)
        const __bf16* ar  = &As[(wm + fr) * LDSS];
        const __bf16* ar2 = &As[(wm + 16 + fr) * LDSS];
        v16bf a0 = frag16(ar  + khalf, ar  + 16 + khalf);
        v16bf a1 = frag16(ar2 + khalf, ar2 + 16 + khalf);
        const __bf16* br  = &Bs[(wn + fr) * LDSS];
        const __bf16* br2 = &Bs[(wn + 16 + fr) * LDSS];
        v16bf b0 = frag16(br  + khalf, br  + 16 + khalf);
        v16bf b1 = frag16(br2 + khalf, br2 + 16 + khalf);

        acc[0][0] = __builtin_amdgcn_wmma_f32_16x16x32_bf16(
            false, a0, false, b0, (short)0, acc[0][0], false, false);
        acc[0][1] = __builtin_amdgcn_wmma_f32_16x16x32_bf16(
            false, a0, false, b1, (short)0, acc[0][1], false, false);
        acc[1][0] = __builtin_amdgcn_wmma_f32_16x16x32_bf16(
            false, a1, false, b0, (short)0, acc[1][0], false, false);
        acc[1][1] = __builtin_amdgcn_wmma_f32_16x16x32_bf16(
            false, a1, false, b1, (short)0, acc[1][1], false, false);
        __syncthreads();
    }

    // ---- store C (ISA 32-bit C/D layout: VGPR r -> M = r + 8*(lane>>4)) ----
    const int rhalf = (lane >> 4) * 8;
    const int cl    = lane & 15;
    #pragma unroll
    for (int mi = 0; mi < 2; ++mi) {
        #pragma unroll
        for (int r = 0; r < 8; ++r) {
            int gr = m0 + wm + mi * 16 + rhalf + r;
            if (gr >= M) continue;
            #pragma unroll
            for (int ni = 0; ni < 2; ++ni) {
                int gn = n0 + wn + ni * 16 + cl;
                if (gn < N) {
                    float v = acc[mi][ni][r];
                    if (bias) v += bias[gn];
                    C[(long)gr * ldc + gn] = activate(v, act);
                }
            }
        }
    }
}

// ---------------------------------------------------------------------------
// Small helper kernels
// ---------------------------------------------------------------------------
__global__ void fill_k(float* p, float v, long n) {
    long i = (long)blockIdx.x * 256 + threadIdx.x;
    if (i < n) p[i] = v;
}

__global__ void degree_k(const int* __restrict__ dst, int E, float* deg) {
    int e = blockIdx.x * 256 + threadIdx.x;
    if (e < E) atomicAdd(&deg[dst[e]], 1.0f);
}

__global__ void rsqrt_k(float* d, int n) {
    int i = blockIdx.x * 256 + threadIdx.x;
    if (i < n) d[i] = rsqrtf(d[i]);
}

// out[i, f] = h[i, f] * inv[i]^2   (self-loop term; F = 256)
__global__ void selfinit_k(const float* __restrict__ h,
                           const float* __restrict__ inv,
                           float* __restrict__ out) {
    int i = blockIdx.x, f = threadIdx.x;
    float s = inv[i] * inv[i];
    out[(long)i * 256 + f] = h[(long)i * 256 + f] * s;
}

// out[dst, f] += h[src, f] * inv[src] * inv[dst]   (one block per edge, F=256)
__global__ void scatter_k(const float* __restrict__ h,
                          const int* __restrict__ src,
                          const int* __restrict__ dst,
                          const float* __restrict__ inv,
                          float* __restrict__ out) {
    int e = blockIdx.x, f = threadIdx.x;
    int s = src[e], d = dst[e];
    float wgt = inv[s] * inv[d];
    atomicAdd(&out[(long)d * 256 + f], h[(long)s * 256 + f] * wgt);
}

__global__ void bias_act_k(float* x, const float* __restrict__ bias,
                           long total, int F, int act) {
    long i = (long)blockIdx.x * 256 + threadIdx.x;
    if (i < total) {
        int f = (int)(i % F);
        x[i] = activate(x[i] + bias[f], act);
    }
}

// per-graph max over 50 nodes, then max with 0 (F = 256)
__global__ void segmax_k(const float* __restrict__ h, float* __restrict__ pooled) {
    int b = blockIdx.x, f = threadIdx.x;
    const float* base = h + (long)b * NODES_PER_GRAPH * 256 + f;
    float m = -3.0e38f;
    #pragma unroll 5
    for (int i = 0; i < NODES_PER_GRAPH; ++i) m = fmaxf(m, base[(long)i * 256]);
    pooled[(long)b * 256 + f] = fmaxf(m, 0.0f);
}

__global__ void copy_cols_k(const float* __restrict__ src, float* __restrict__ dst,
                            long n, int F, int ldd, int coff) {
    long i = (long)blockIdx.x * 256 + threadIdx.x;
    if (i < n * F) {
        long r = i / F; int c = (int)(i % F);
        dst[r * ldd + coff + c] = src[i];
    }
}

// pcls = w0*P0 + w1*P1 + w2*P2 + b
__global__ void combine_k(const float* __restrict__ P0, const float* __restrict__ P1,
                          const float* __restrict__ P2, const float* __restrict__ Wv,
                          const float* __restrict__ bv, float* __restrict__ out, long n) {
    long i = (long)blockIdx.x * 256 + threadIdx.x;
    if (i < n) out[i] = Wv[0] * P0[i] + Wv[1] * P1[i] + Wv[2] * P2[i] + bv[0];
}

// ---------------------------------------------------------------------------
// Host-side helpers
// ---------------------------------------------------------------------------
static void gemm(hipStream_t s, const float* A, const float* B, const float* bias,
                 float* C, int M, int N, int K, int ldc, int transB, int act) {
    dim3 g((N + TN - 1) / TN, (M + TM - 1) / TM);
    gemm_wmma_bf16<<<g, 256, 0, s>>>(A, B, bias, C, M, N, K, ldc, transB, act);
}

extern "C" void kernel_launch(void* const* d_in, const int* in_sizes, int n_in,
                              void* d_out, int out_size, void* d_ws, size_t ws_size,
                              hipStream_t stream) {
    (void)in_sizes; (void)n_in; (void)out_size; (void)ws_size;
    #define IN(i) ((const float*)d_in[i])
    const float* mol_x = IN(0);
    const float* txt   = IN(1);
    const float* desc  = IN(2);
    const float* sef   = IN(3);
    const int* edge_src    = (const int*)d_in[78];
    const int* edge_dst    = (const int*)d_in[79];
    const int* se_edge_src = (const int*)d_in[81];
    const int* se_edge_dst = (const int*)d_in[82];

    float* out = (float*)d_out;                 // preg [2048*994] then pcls [2048*994]
    float* ws  = (float*)d_ws;
    size_t off = 0;
    auto alloc = [&](size_t n) { float* p = ws + off; off += n; return p; };
    auto blocks = [](long n) { return (unsigned)((n + 255) / 256); };

    // ----- SE graph degree / inv -----
    float* inv_se = alloc(NUM_SE);
    fill_k<<<blocks(NUM_SE), 256, 0, stream>>>(inv_se, 1.0f, NUM_SE);
    degree_k<<<blocks(NUM_SE_EDGES), 256, 0, stream>>>(se_edge_dst, NUM_SE_EDGES, inv_se);
    rsqrt_k<<<blocks(NUM_SE), 256, 0, stream>>>(inv_se, NUM_SE);

    // ----- SE dense encoder: tanh(relu(sef@W1+b1)@W2+b2) -> ini[:, :128] -----
    float* ini = alloc((size_t)NUM_SE * 384);
    float* t1  = alloc((size_t)NUM_SE * 128);
    gemm(stream, sef, IN(4), IN(5), t1, NUM_SE, 128, SE_FEAT, 128, 0, 1);
    gemm(stream, t1, IN(6), IN(7), ini, NUM_SE, 128, 128, 384, 0, 2);

    // ----- SE GCN stack (3 layers, F=256) -> ini[:, 128:384] -----
    float* seA = alloc((size_t)NUM_SE * 256);
    float* seB = alloc((size_t)NUM_SE * 256);
    const float* se_in = sef; int se_k = SE_FEAT;
    int se_w[3] = {62, 64, 66};
    for (int L = 0; L < 3; ++L) {
        gemm(stream, se_in, IN(se_w[L]), nullptr, seA, NUM_SE, 256, se_k, 256, 0, 0);
        selfinit_k<<<NUM_SE, 256, 0, stream>>>(seA, inv_se, seB);
        scatter_k<<<NUM_SE_EDGES, 256, 0, stream>>>(seA, se_edge_src, se_edge_dst, inv_se, seB);
        bias_act_k<<<blocks((long)NUM_SE * 256), 256, 0, stream>>>(
            seB, IN(se_w[L] + 1), (long)NUM_SE * 256, 256, 3);
        se_in = seB; se_k = 256;
    }
    copy_cols_k<<<blocks((long)NUM_SE * 256), 256, 0, stream>>>(seB, ini, NUM_SE, 256, 384, 128);

    // ----- four MLP3 heads on ini [994,384] -> st/st2/st8/st10 [994,256] -----
    float* mh1  = alloc((size_t)NUM_SE * 384);
    float* mh2  = alloc((size_t)NUM_SE * 256);
    float* st   = alloc((size_t)NUM_SE * 256);
    float* st2  = alloc((size_t)NUM_SE * 256);
    float* st8  = alloc((size_t)NUM_SE * 256);
    float* st10 = alloc((size_t)NUM_SE * 256);
    float* st_out[4] = {st, st2, st8, st10};
    int st_base[4] = {8, 14, 20, 26};
    for (int i = 0; i < 4; ++i) {
        int p = st_base[i];
        gemm(stream, ini, IN(p),     IN(p + 1), mh1,       NUM_SE, 384, 384, 384, 0, 1);
        gemm(stream, mh1, IN(p + 2), IN(p + 3), mh2,       NUM_SE, 256, 384, 256, 0, 1);
        gemm(stream, mh2, IN(p + 4), IN(p + 5), st_out[i], NUM_SE, 256, 256, 256, 0, 2);
    }

    // ----- molecular graph degree / inv -----
    float* inv_mol = alloc(NUM_NODES);
    fill_k<<<blocks(NUM_NODES), 256, 0, stream>>>(inv_mol, 1.0f, NUM_NODES);
    degree_k<<<blocks(NUM_EDGES), 256, 0, stream>>>(edge_dst, NUM_EDGES, inv_mol);
    rsqrt_k<<<blocks(NUM_NODES), 256, 0, stream>>>(inv_mol, NUM_NODES);

    // ----- molecular GCN stack (3 layers, F=256), ping-pong bufA/bufB -----
    float* bufA = alloc((size_t)NUM_NODES * 256);
    float* bufB = alloc((size_t)NUM_NODES * 256);
    const float* mol_in = mol_x; int mol_k = 110;
    int mol_w[3] = {56, 58, 60};
    for (int L = 0; L < 3; ++L) {
        gemm(stream, mol_in, IN(mol_w[L]), nullptr, bufA, NUM_NODES, 256, mol_k, 256, 0, 0);
        selfinit_k<<<NUM_NODES, 256, 0, stream>>>(bufA, inv_mol, bufB);
        scatter_k<<<NUM_EDGES, 256, 0, stream>>>(bufA, edge_src, edge_dst, inv_mol, bufB);
        bias_act_k<<<blocks((long)NUM_NODES * 256), 256, 0, stream>>>(
            bufB, IN(mol_w[L] + 1), (long)NUM_NODES * 256, 256, 3);
        mol_in = bufB; mol_k = 256;
    }

    // ----- per-graph max pool (clamped at 0) -----
    float* pooled = alloc((size_t)NUM_GRAPHS * 256);
    segmax_k<<<NUM_GRAPHS, 256, 0, stream>>>(bufB, pooled);

    // ----- gfc / gfr (gfr goes straight into dar_in cols 0:256) -----
    float* dar_in = alloc((size_t)NUM_GRAPHS * 768);
    float* gfc    = alloc((size_t)NUM_GRAPHS * 256);
    gemm(stream, pooled, IN(68), IN(69), gfc,    NUM_GRAPHS, 256, 256, 256, 0, 0);
    gemm(stream, pooled, IN(70), IN(71), dar_in, NUM_GRAPHS, 256, 256, 768, 0, 0);

    // ----- text MLPs [750,750,512,256]: cls -> txt_c, reg -> dar_in cols 256:512
    float* th1   = alloc((size_t)NUM_GRAPHS * 750);
    float* th2   = alloc((size_t)NUM_GRAPHS * 512);
    float* txt_c = alloc((size_t)NUM_GRAPHS * 256);
    gemm(stream, txt, IN(32), IN(33), th1,   NUM_GRAPHS, 750, 750, 750, 0, 1);
    gemm(stream, th1, IN(34), IN(35), th2,   NUM_GRAPHS, 512, 750, 512, 0, 1);
    gemm(stream, th2, IN(36), IN(37), txt_c, NUM_GRAPHS, 256, 512, 256, 0, 2);
    gemm(stream, txt, IN(38), IN(39), th1,   NUM_GRAPHS, 750, 750, 750, 0, 1);
    gemm(stream, th1, IN(40), IN(41), th2,   NUM_GRAPHS, 512, 750, 512, 0, 1);
    gemm(stream, th2, IN(42), IN(43), dar_in + 256, NUM_GRAPHS, 256, 512, 768, 0, 2);

    // ----- desc MLPs [384,384,384,256]: cls -> dsc_c, reg -> dar_in cols 512:768
    float* dh1   = alloc((size_t)NUM_GRAPHS * 384);
    float* dh2   = alloc((size_t)NUM_GRAPHS * 384);
    float* dsc_c = alloc((size_t)NUM_GRAPHS * 256);
    gemm(stream, desc, IN(44), IN(45), dh1,   NUM_GRAPHS, 384, 384, 384, 0, 1);
    gemm(stream, dh1,  IN(46), IN(47), dh2,   NUM_GRAPHS, 384, 384, 384, 0, 1);
    gemm(stream, dh2,  IN(48), IN(49), dsc_c, NUM_GRAPHS, 256, 384, 256, 0, 2);
    gemm(stream, desc, IN(50), IN(51), dh1,   NUM_GRAPHS, 384, 384, 384, 0, 1);
    gemm(stream, dh1,  IN(52), IN(53), dh2,   NUM_GRAPHS, 384, 384, 384, 0, 1);
    gemm(stream, dh2,  IN(54), IN(55), dar_in + 512, NUM_GRAPHS, 256, 384, 768, 0, 2);

    // ----- ser = st @ se_reg_layer + b; dar = dar_in @ drug_reg_layer + b -----
    float* ser = alloc((size_t)NUM_SE * 256);
    float* dar = alloc((size_t)NUM_GRAPHS * 256);
    gemm(stream, st,     IN(74), IN(75), ser, NUM_SE,     256, 256, 256, 0, 0);
    gemm(stream, dar_in, IN(72), IN(73), dar, NUM_GRAPHS, 256, 768, 256, 0, 0);

    // ----- preg = dar @ ser^T -> out[:2048*994] -----
    gemm(stream, dar, ser, nullptr, out, NUM_GRAPHS, NUM_SE, 256, NUM_SE, 1, 0);

    // ----- three similarity views + view_mlp -> pcls -----
    float* P0 = alloc((size_t)NUM_GRAPHS * NUM_SE);
    float* P1 = alloc((size_t)NUM_GRAPHS * NUM_SE);
    float* P2 = alloc((size_t)NUM_GRAPHS * NUM_SE);
    gemm(stream, gfc,   st2,  nullptr, P0, NUM_GRAPHS, NUM_SE, 256, NUM_SE, 1, 0);
    gemm(stream, txt_c, st8,  nullptr, P1, NUM_GRAPHS, NUM_SE, 256, NUM_SE, 1, 0);
    gemm(stream, dsc_c, st10, nullptr, P2, NUM_GRAPHS, NUM_SE, 256, NUM_SE, 1, 0);
    long ncls = (long)NUM_GRAPHS * NUM_SE;
    combine_k<<<blocks(ncls), 256, 0, stream>>>(P0, P1, P2, IN(76), IN(77),
                                                out + ncls, ncls);
    #undef IN
}
